// DASTNet_56848187130530
// MI455X (gfx1250) — matrix operand order repaired and checked by reading, not compiled, not run on hardware
//
#include <hip/hip_runtime.h>
#include <hip/hip_bf16.h>

typedef __attribute__((ext_vector_type(16))) _Float16 v16h;
typedef __attribute__((ext_vector_type(8)))  _Float16 v8h;
typedef __attribute__((ext_vector_type(8)))  float    v8f;

#define B_   64
#define S_   12
#define N4_  307
#define N7_  883
#define N8_  170
#define IN_  128
#define H_   256
#define E_   256
#define BN_  (B_ * N4_)   /* 19648, multiple of 16 */
#define PRE_ 12

// ---------------------------------------------------------------------------
// Elementwise / packing kernels (fp32 math, f16 conversion for WMMA operands)
// ---------------------------------------------------------------------------

__global__ void k_zero_f32(float* p, long n) {
  long t = (long)blockIdx.x * blockDim.x + threadIdx.x;
  if (t < n) p[t] = 0.0f;
}

// f32 (M x K) row-major -> f16 (M x Kpad) row-major, zero pad K..Kpad
__global__ void k_f32_to_f16_pad(const float* __restrict__ s, _Float16* __restrict__ d,
                                 int M, int K, int Kpad) {
  long t = (long)blockIdx.x * blockDim.x + threadIdx.x;
  long n = (long)M * Kpad;
  if (t >= n) return;
  int r = (int)(t / Kpad), c = (int)(t % Kpad);
  d[t] = (c < K) ? (_Float16)s[(long)r * K + c] : (_Float16)0.0f;
}

// Pack f32 B (K x Ncols, row-major) into WMMA B-fragment layout:
//   Bp[((nt*Ktiles + kt)*32 + lane)*16 + i] = B[kt*32 + (lane>>4)*16 + i][nt*16 + (lane&15)]
// zero-padded outside K/Ncols. (contiguous k order; A loads in gemm_f16_kernel use the same)
__global__ void k_pack_b(const float* __restrict__ Bsrc, _Float16* __restrict__ Bp,
                         int K, int Ncols, int Ktiles, int Ntiles) {
  int t = blockIdx.x * blockDim.x + threadIdx.x;
  int tot = Ntiles * Ktiles * 32;
  if (t >= tot) return;
  int lane = t & 31;
  int kt = (t >> 5) % Ktiles;
  int nt = (t >> 5) / Ktiles;
  int kbase = kt * 32 + (lane >> 4) * 16;
  int col = nt * 16 + (lane & 15);
  v16h v;
#pragma unroll
  for (int i = 0; i < 16; ++i) {
    int k = kbase + i;
    float x = (k < K && col < Ncols) ? Bsrc[(long)k * Ncols + col] : 0.0f;
    v[i] = (_Float16)x;
  }
  *(v16h*)(Bp + (long)t * 16) = v;
}

__global__ void k_rowsum(const float* __restrict__ adj, float* __restrict__ deg, int N) {
  int r = blockIdx.x * blockDim.x + threadIdx.x;
  if (r >= N) return;
  float s = 0.0f;
  for (int c = 0; c < N; ++c) s += adj[(long)r * N + c];
  deg[r] = s;
}

// g = pooled/deg + eps*h1   (in-place on pooled), cols = 256
__global__ void k_gin(float* __restrict__ g, const float* __restrict__ h1,
                      const float* __restrict__ deg, const float* __restrict__ eps, int N) {
  long t = (long)blockIdx.x * blockDim.x + threadIdx.x;
  long n = (long)N * E_;
  if (t >= n) return;
  int r = (int)(t >> 8);
  g[t] = g[t] / deg[r] + eps[0] * h1[t];
}

// Per-column batch stats over N rows (256 cols, one block of 256 threads; coalesced)
__global__ void k_bn_stats(const float* __restrict__ g, float* __restrict__ mu,
                           float* __restrict__ rst, int N) {
  int c = threadIdx.x;
  float s1 = 0.0f, s2 = 0.0f;
  for (int r = 0; r < N; ++r) {
    float x = g[(long)r * E_ + c];
    s1 += x; s2 += x * x;
  }
  float inv = 1.0f / (float)N;
  float m = s1 * inv;
  float v = s2 * inv - m * m;
  mu[c] = m;
  rst[c] = rsqrtf(v + 1e-5f);
}

__global__ void k_bn_apply_f16(const float* __restrict__ g, const float* __restrict__ mu,
                               const float* __restrict__ rst, const float* __restrict__ gam,
                               const float* __restrict__ bet, _Float16* __restrict__ d, long n) {
  long t = (long)blockIdx.x * blockDim.x + threadIdx.x;
  if (t >= n) return;
  int c = (int)(t & 255);
  d[t] = (_Float16)((g[t] - mu[c]) * rst[c] * gam[c] + bet[c]);
}

// cat1 = [x_t | h | zeros]  f16, (BN x 288)
__global__ void k_cat1(const float* __restrict__ feat, const float* __restrict__ h,
                       _Float16* __restrict__ cat, int t) {
  long i = (long)blockIdx.x * blockDim.x + threadIdx.x;
  const long n = (long)BN_ * 288;
  if (i >= n) return;
  int row = (int)(i / 288), c = (int)(i % 288);
  float v = 0.0f;
  if (c == 0) {
    int b = row / N4_, nn = row % N4_;
    v = feat[((long)b * S_ + t) * N4_ + nn];
  } else if (c <= H_) {
    v = h[(long)row * H_ + (c - 1)];
  }
  cat[i] = (_Float16)v;
}

// cat2 = [x_t | r .* h | zeros]; r taken from FLAT split of sigmoid gates,
// per-batch flat index i = n*H + hcol  (matches jnp.split on (B, N*2H) axis=1)
__global__ void k_cat2(const float* __restrict__ feat, const float* __restrict__ h,
                       const float* __restrict__ gates, _Float16* __restrict__ cat, int t) {
  long i = (long)blockIdx.x * blockDim.x + threadIdx.x;
  const long n = (long)BN_ * 288;
  if (i >= n) return;
  int row = (int)(i / 288), c = (int)(i % 288);
  int b = row / N4_, nn = row % N4_;
  float v = 0.0f;
  if (c == 0) {
    v = feat[((long)b * S_ + t) * N4_ + nn];
  } else if (c <= H_) {
    int hc = c - 1;
    float rr = gates[(long)b * (N4_ * 2 * H_) + (long)nn * H_ + hc];
    v = rr * h[(long)row * H_ + hc];
  }
  cat[i] = (_Float16)v;
}

// nh0 = u*h + (1-u)*c ; u from second flat half of gates; write into cat3[:, 0:256]
__global__ void k_gru_combine(const float* __restrict__ gates, const float* __restrict__ h,
                              const float* __restrict__ cv, _Float16* __restrict__ cat3) {
  long i = (long)blockIdx.x * blockDim.x + threadIdx.x;
  const long n = (long)BN_ * H_;
  if (i >= n) return;
  int row = (int)(i >> 8), hc = (int)(i & 255);
  int b = row / N4_, nn = row % N4_;
  float u = gates[(long)b * (N4_ * 2 * H_) + (long)N4_ * H_ + (long)nn * H_ + hc];
  float nh = u * h[i] + (1.0f - u) * cv[i];
  cat3[(long)row * (2 * H_) + hc] = (_Float16)nh;
}

// cat3[:, 256:512] = fp_small[n] (broadcast across batch); done once
__global__ void k_fp_half(const float* __restrict__ fp, _Float16* __restrict__ cat3) {
  long i = (long)blockIdx.x * blockDim.x + threadIdx.x;
  const long n = (long)BN_ * E_;
  if (i >= n) return;
  int row = (int)(i >> 8), c = (int)(i & 255);
  int nn = row % N4_;
  cat3[(long)row * 512 + 256 + c] = (_Float16)fp[(long)nn * E_ + c];
}

// ---------------------------------------------------------------------------
// WMMA GEMM: C[M,Ncols] = A[M,Kpad(f16)] @ Bpacked + bias, optional activation.
// One wave -> 16 x (16*NT) tile; K-loop in steps of 32 via v_wmma_f32_16x16x32_f16.
// A layout (16-bit A 16x32, contiguous k): lane = row(&15), kgroup = lane>>4,
//   elems 0..15 -> K = kt*32 + kgroup*16 + i (same order as packed B).
// ---------------------------------------------------------------------------
template <int NT>
__global__ void gemm_f16_kernel(const _Float16* __restrict__ A, const _Float16* __restrict__ Bp,
                                const float* __restrict__ bias, int bias_scalar,
                                float* __restrict__ C,
                                int M, int Kpad, int Ktiles, int Ntiles, int NtG,
                                int Ncols, int ldc, int op, int tasks) {
  int wave = blockIdx.x * (blockDim.x >> 5) + (threadIdx.x >> 5);
  if (wave >= tasks) return;                      // uniform per-wave exit; EXEC stays full
  int mtile = wave / NtG;
  int ntile0 = (wave % NtG) * NT;
  int lane = threadIdx.x & 31;

  int rowA = mtile * 16 + (lane & 15);
  if (rowA >= M) rowA = M - 1;                    // clamp (edge rows masked on store)
  const _Float16* arow = A + (long)rowA * Kpad + ((lane >> 4) * 16);

  v8f acc[NT];
#pragma unroll
  for (int j = 0; j < NT; ++j)
    acc[j] = (v8f){0.f, 0.f, 0.f, 0.f, 0.f, 0.f, 0.f, 0.f};

  for (int kt = 0; kt < Ktiles; ++kt) {
    v8h alo = *(const v8h*)(arow + kt * 32);
    v8h ahi = *(const v8h*)(arow + kt * 32 + 8);
    v16h a = __builtin_shufflevector(alo, ahi, 0, 1, 2, 3, 4, 5, 6, 7,
                                     8, 9, 10, 11, 12, 13, 14, 15);
#pragma unroll
    for (int j = 0; j < NT; ++j) {
      const _Float16* bptr = Bp + ((long)(ntile0 + j) * Ktiles + kt) * (32 * 16) + lane * 16;
      v16h b = *(const v16h*)bptr;
      acc[j] = __builtin_amdgcn_wmma_f32_16x16x32_f16(false, a, false, b,
                                                      (short)0, acc[j], false, false);
    }
  }

  int rBase = mtile * 16 + (lane >> 4) * 8;       // C layout: VGPR v -> M = kgroup*8+v
  int cl = lane & 15;
#pragma unroll
  for (int j = 0; j < NT; ++j) {
    int cc = (ntile0 + j) * 16 + cl;
    if (cc >= Ncols) continue;
    float bv = bias ? (bias_scalar ? bias[0] : bias[cc]) : 0.0f;
#pragma unroll
    for (int v = 0; v < 8; ++v) {
      int rr = rBase + v;
      if (rr < M) {
        float x = acc[j][v] + bv;
        if (op == 1) x = 1.0f / (1.0f + expf(-x));
        else if (op == 2) x = tanhf(x);
        C[(long)rr * ldc + cc] = x;
      }
    }
  }
}

static inline void launch_gemm(int NT, const _Float16* A, int Kpad, int Ktiles,
                               const _Float16* Bp, const float* bias, int bias_scalar,
                               float* C, int M, int Ncols, int ldc, int op,
                               hipStream_t stream) {
  int Mtiles = (M + 15) / 16;
  int Ntiles = (Ncols + 15) / 16;
  int NtG = (NT == 4) ? (Ntiles / 4) : Ntiles;
  if (NtG < 1) NtG = 1;
  int tasks = Mtiles * NtG;
  int blocks = (tasks + 7) / 8;                   // 8 waves (256 thr) per block
  if (NT == 4)
    gemm_f16_kernel<4><<<blocks, 256, 0, stream>>>(A, Bp, bias, bias_scalar, C, M, Kpad,
                                                   Ktiles, Ntiles, NtG, Ncols, ldc, op, tasks);
  else
    gemm_f16_kernel<1><<<blocks, 256, 0, stream>>>(A, Bp, bias, bias_scalar, C, M, Kpad,
                                                   Ktiles, Ntiles, NtG, Ncols, ldc, op, tasks);
}

// ---------------------------------------------------------------------------
extern "C" void kernel_launch(void* const* d_in, const int* in_sizes, int n_in,
                              void* d_out, int out_size, void* d_ws, size_t ws_size,
                              hipStream_t stream) {
  (void)in_sizes; (void)n_in; (void)out_size; (void)ws_size;

  const float* vec4 = (const float*)d_in[0];
  const float* vec7 = (const float*)d_in[1];
  const float* vec8 = (const float*)d_in[2];
  const float* feat = (const float*)d_in[3];
  const float* adj4 = (const float*)d_in[4];
  const float* adj7 = (const float*)d_in[5];
  const float* adj8 = (const float*)d_in[6];

  struct Ex { const float *W1, *b1, *W2, *b2, *g, *bt, *eps; };
  auto exP = [&](int base) {
    Ex e;
    e.W1 = (const float*)d_in[base + 0]; e.b1 = (const float*)d_in[base + 1];
    e.W2 = (const float*)d_in[base + 2]; e.b2 = (const float*)d_in[base + 3];
    e.g  = (const float*)d_in[base + 4]; e.bt = (const float*)d_in[base + 5];
    e.eps = (const float*)d_in[base + 6];
    return e;
  };
  Ex ex4 = exP(7), ex7 = exP(14), ex8 = exP(21);
  const float* gruW1 = (const float*)d_in[28];
  const float* grub1 = (const float*)d_in[29];
  const float* gruW2 = (const float*)d_in[30];
  const float* grub2 = (const float*)d_in[31];
  const float* featW = (const float*)d_in[32];
  const float* featb = (const float*)d_in[33];   // scalar
  const float* linW  = (const float*)d_in[34];
  const float* linb  = (const float*)d_in[35];
  const float* outW  = (const float*)d_in[36];
  const float* outb  = (const float*)d_in[37];

  // d_out layout: pred | f4 | f7 | f8 (all fp32, row-major)
  float* pred = (float*)d_out;
  float* f4o = pred + (long)B_ * N4_ * PRE_;
  float* f7o = f4o + (long)N4_ * E_;
  float* f8o = f7o + (long)N7_ * E_;

  // ---- workspace carve-out (256B aligned slots) ----
  char* ws = (char*)d_ws;
  size_t off = 0;
  auto alloc = [&](size_t bytes) -> char* {
    off = (off + 255) & ~(size_t)255;
    char* p = ws + off; off += bytes; return p;
  };
  _Float16* pW1[3]; _Float16* pW2[3];
  for (int i = 0; i < 3; ++i) {
    pW1[i] = (_Float16*)alloc((size_t)4 * 16 * 512 * 2);   // K=128 -> 4 ktiles
    pW2[i] = (_Float16*)alloc((size_t)8 * 16 * 512 * 2);   // K=256 -> 8 ktiles
  }
  _Float16* pGW1 = (_Float16*)alloc((size_t)9 * 32 * 512 * 2);  // 288x512
  _Float16* pGW2 = (_Float16*)alloc((size_t)9 * 16 * 512 * 2);  // 288x256
  _Float16* pFW  = (_Float16*)alloc((size_t)8 * 16 * 512 * 2);  // 256x256
  _Float16* pLW  = (_Float16*)alloc((size_t)16 * 16 * 512 * 2); // 512x256
  _Float16* pOW  = (_Float16*)alloc((size_t)8 * 1 * 512 * 2);   // 256x16(12)
  float* hbuf  = (float*)alloc((size_t)BN_ * H_ * 4);
  float* gates = (float*)alloc((size_t)BN_ * 2 * H_ * 4);
  float* cbuf  = (float*)alloc((size_t)BN_ * H_ * 4);
  float* fps   = (float*)alloc((size_t)N4_ * E_ * 4);
  _Float16* cat1 = (_Float16*)alloc((size_t)BN_ * 288 * 2);  // also reused as hA (BN x 256)
  _Float16* cat3 = (_Float16*)alloc((size_t)BN_ * 512 * 2);
  _Float16* vecA = (_Float16*)alloc((size_t)N7_ * IN_ * 2);
  _Float16* adjA = (_Float16*)alloc((size_t)N7_ * 896 * 2);
  float* h1 = (float*)alloc((size_t)N7_ * E_ * 4);
  _Float16* h1p = (_Float16*)alloc((size_t)28 * 16 * 512 * 2);
  float* pooled = (float*)alloc((size_t)N7_ * E_ * 4);
  _Float16* hnA = (_Float16*)alloc((size_t)N7_ * E_ * 2);
  _Float16* f4A = (_Float16*)alloc((size_t)N4_ * E_ * 2);
  float* deg = (float*)alloc((size_t)N7_ * 4);
  float* mu  = (float*)alloc(256 * 4);
  float* rst = (float*)alloc(256 * 4);

  auto nb = [](long n) { return (unsigned)((n + 255) / 256); };
  auto pack = [&](const float* src, _Float16* dst, int K, int Ncols, int Kt, int Nt) {
    long tot = (long)Nt * Kt * 32;
    k_pack_b<<<nb(tot), 256, 0, stream>>>(src, dst, K, Ncols, Kt, Nt);
  };

  // ---- pack all weights once ----
  pack(ex4.W1, pW1[0], IN_, H_, 4, 16);  pack(ex4.W2, pW2[0], H_, E_, 8, 16);
  pack(ex7.W1, pW1[1], IN_, H_, 4, 16);  pack(ex7.W2, pW2[1], H_, E_, 8, 16);
  pack(ex8.W1, pW1[2], IN_, H_, 4, 16);  pack(ex8.W2, pW2[2], H_, E_, 8, 16);
  pack(gruW1, pGW1, H_ + 1, 2 * H_, 9, 32);
  pack(gruW2, pGW2, H_ + 1, H_, 9, 16);
  pack(featW, pFW, E_, E_, 8, 16);
  pack(linW, pLW, H_ + E_, H_, 16, 16);
  pack(outW, pOW, H_, PRE_, 8, 1);

  // ---- extractors ----
  auto extract = [&](int N, int Npad, const float* vec, const float* adj,
                     const Ex& e, const _Float16* W1p, const _Float16* W2p, float* fout) {
    k_f32_to_f16_pad<<<nb((long)N * IN_), 256, 0, stream>>>(vec, vecA, N, IN_, IN_);
    launch_gemm(4, vecA, IN_, 4, W1p, e.b1, 0, h1, N, H_, H_, 0, stream);
    k_rowsum<<<nb(N), 256, 0, stream>>>(adj, deg, N);
    k_f32_to_f16_pad<<<nb((long)N * Npad), 256, 0, stream>>>(adj, adjA, N, N, Npad);
    pack(h1, h1p, N, E_, Npad / 32, 16);
    launch_gemm(4, adjA, Npad, Npad / 32, h1p, nullptr, 0, pooled, N, E_, E_, 0, stream);
    k_gin<<<nb((long)N * E_), 256, 0, stream>>>(pooled, h1, deg, e.eps, N);
    k_bn_stats<<<1, 256, 0, stream>>>(pooled, mu, rst, N);
    k_bn_apply_f16<<<nb((long)N * E_), 256, 0, stream>>>(pooled, mu, rst, e.g, e.bt, hnA,
                                                         (long)N * E_);
    launch_gemm(4, hnA, E_, 8, W2p, e.b2, 0, fout, N, E_, E_, 0, stream);
  };
  extract(N4_, 320, vec4, adj4, ex4, pW1[0], pW2[0], f4o);
  extract(N7_, 896, vec7, adj7, ex7, pW1[1], pW2[1], f7o);
  extract(N8_, 192, vec8, adj8, ex8, pW1[2], pW2[2], f8o);

  // ---- fp = f4 @ feat_W + feat_b (hoisted, broadcast across batch) ----
  k_f32_to_f16_pad<<<nb((long)N4_ * E_), 256, 0, stream>>>(f4o, f4A, N4_, E_, E_);
  launch_gemm(4, f4A, E_, 8, pFW, featb, 1, fps, N4_, E_, E_, 0, stream);
  k_fp_half<<<nb((long)BN_ * E_), 256, 0, stream>>>(fps, cat3);

  // ---- GRU scan over 12 timesteps ----
  k_zero_f32<<<nb((long)BN_ * H_), 256, 0, stream>>>(hbuf, (long)BN_ * H_);
  for (int t = 0; t < S_; ++t) {
    k_cat1<<<nb((long)BN_ * 288), 256, 0, stream>>>(feat, hbuf, cat1, t);
    launch_gemm(4, cat1, 288, 9, pGW1, grub1, 0, gates, BN_, 2 * H_, 2 * H_, 1, stream); // sigmoid
    k_cat2<<<nb((long)BN_ * 288), 256, 0, stream>>>(feat, hbuf, gates, cat1, t);
    launch_gemm(4, cat1, 288, 9, pGW2, grub2, 0, cbuf, BN_, H_, H_, 2, stream);          // tanh
    k_gru_combine<<<nb((long)BN_ * H_), 256, 0, stream>>>(gates, hbuf, cbuf, cat3);
    launch_gemm(4, cat3, 512, 16, pLW, linb, 0, hbuf, BN_, H_, H_, 0, stream);           // new h
  }

  // ---- prediction head: pred = hT @ out_W + out_b  (12 cols, masked store) ----
  k_f32_to_f16_pad<<<nb((long)BN_ * H_), 256, 0, stream>>>(hbuf, cat1, BN_, H_, H_);
  launch_gemm(1, cat1, H_, 8, pOW, outb, 0, pred, BN_, PRE_, PRE_, 0, stream);
}
